// CharEncTrans_9440338117148
// MI455X (gfx1250) — matrix operand order repaired and checked by reading, not compiled, not run on hardware
//
#include <hip/hip_runtime.h>
#include <hip/hip_bf16.h>

typedef __attribute__((ext_vector_type(16))) _Float16 v16h;
typedef __attribute__((ext_vector_type(8)))  float    v8f;

#define DEVINL __device__ __forceinline__

namespace {
constexpr int kB = 64, kS = 4096, kT = 512, kE = 28, kHD = 7, kFF = 256, kLMAX = 8;
constexpr int kEP   = 32;   // E padded to 32 (WMMA K)
constexpr int kQKVP = 96;   // 3E padded to 96
constexpr int NW    = 4;    // waves per block
constexpr int BLOCK = NW * 32;
constexpr int NPAIR = kB * kT / 2;   // 16384 span-pairs
constexpr int GRID  = NPAIR / NW;    // 4096 blocks
}

// Wave-private LDS RAW fence: stores above must complete before loads below.
DEVINL void lds_fence() {
  __builtin_amdgcn_wave_barrier();
  asm volatile("s_wait_dscnt 0x0" ::: "memory");
  __builtin_amdgcn_wave_barrier();
}

// A fragment (16x32 f16, row-major, stride 32 halves).
// ISA 16-bit A layout: lane L -> row L%16; elem i<8 -> K = i + 8*(L/16);
// elem i>=8 -> K = i + 8 + 8*(L/16).
DEVINL v16h frag_a(const _Float16* tile, int lane) {
  const int row = lane & 15, hf = lane >> 4;
  const _Float16* p0 = tile + row * kEP + 8 * hf;
  const _Float16* p1 = p0 + 16;
  v16h a;
#pragma unroll
  for (int i = 0; i < 8; ++i) { a[i] = p0[i]; a[i + 8] = p1[i]; }
  return a;
}

// B fragment: lane L -> column N = L%16; elem i<8 -> K = i + 8*(L/16);
// elem i>=8 -> K = i + 8 + 8*(L/16).
// colbase points at W[n][k0] (K contiguous in memory).
DEVINL v16h frag_b(const _Float16* colbase, int hf) {
  const _Float16* p0 = colbase + 8 * hf;
  const _Float16* p1 = p0 + 16;
  v16h b;
#pragma unroll
  for (int i = 0; i < 8; ++i) { b[i] = p0[i]; b[i + 8] = p1[i]; }
  return b;
}

DEVINL v8f wmma_f16(v16h a, v16h b, v8f c) {
  return __builtin_amdgcn_wmma_f32_16x16x32_f16(false, a, false, b, (short)0, c,
                                                false, false);
}

// Row-wise layernorm over 28 valid cols. C layout: lane holds col c (+16),
// rows r + 8*hf; a logical row is spread over the 16 lanes of one half-wave,
// so shfl_xor with masks 1/2/4/8 reduces exactly within a row.
DEVINL void layer_norm28(const float* y0, const float* y1, const float* g,
                         const float* bp, int c, float* z0, float* z1) {
#pragma unroll
  for (int r = 0; r < 8; ++r) {
    float ps = y0[r] + y1[r];
    float pq = y0[r] * y0[r] + y1[r] * y1[r];
#pragma unroll
    for (int m = 1; m < 16; m <<= 1) {
      ps += __shfl_xor(ps, m, 32);
      pq += __shfl_xor(pq, m, 32);
    }
    const float mean = ps * (1.f / 28.f);
    const float var  = pq * (1.f / 28.f) - mean * mean;
    const float inv  = rsqrtf(var + 1e-5f);
    z0[r] = (y0[r] - mean) * inv * g[c] + bp[c];
    z1[r] = (c < 12) ? ((y1[r] - mean) * inv * g[16 + c] + bp[16 + c]) : 0.f;
  }
}

__global__ __launch_bounds__(BLOCK) void span_encoder_kernel(
    const float* __restrict__ emb, const int* __restrict__ span_lengths,
    const int* __restrict__ num_spans,
    const float* __restrict__ in_proj_w, const float* __restrict__ in_proj_b,
    const float* __restrict__ out_proj_w, const float* __restrict__ out_proj_b,
    const float* __restrict__ ln1_g, const float* __restrict__ ln1_b,
    const float* __restrict__ lin1_w, const float* __restrict__ lin1_b,
    const float* __restrict__ lin2_w, const float* __restrict__ lin2_b,
    const float* __restrict__ ln2_g, const float* __restrict__ ln2_b,
    const float* __restrict__ pad_token, float* __restrict__ out) {
  // ---- block-shared weights (f16, zero-padded) ----
  __shared__ _Float16 sWqkv[kQKVP * kEP];   // [n][k], n<84 valid
  __shared__ _Float16 sWout[kEP * kEP];     // [n][k], n<28 valid
  __shared__ _Float16 sW1[kFF * kEP];       // [n][k]
  __shared__ _Float16 sW2[kEP * kFF];       // [e][f] (B[k=f][n=e])
  __shared__ float sBqkv[kQKVP], sBout[kEP], sB1[kFF], sB2[kEP];
  __shared__ float sLn1g[kEP], sLn1b[kEP], sLn2g[kEP], sLn2b[kEP], sPad[kEP];
  // ---- wave-private staging ----
  __shared__ _Float16 sQKV[NW][16 * kQKVP]; // f16 qkv per pair
  __shared__ _Float16 sT[NW][16 * kEP];     // x / attn-out / ln1-out / relu(h)

  const int tid = threadIdx.x;
  for (int i = tid; i < kQKVP * kEP; i += BLOCK) {
    int n = i >> 5, k = i & 31;
    sWqkv[i] = (_Float16)((n < 84 && k < kE) ? in_proj_w[n * kE + k] : 0.f);
  }
  for (int i = tid; i < kEP * kEP; i += BLOCK) {
    int n = i >> 5, k = i & 31;
    sWout[i] = (_Float16)((n < kE && k < kE) ? out_proj_w[n * kE + k] : 0.f);
  }
  for (int i = tid; i < kFF * kEP; i += BLOCK) {
    int n = i >> 5, k = i & 31;
    sW1[i] = (_Float16)((k < kE) ? lin1_w[n * kE + k] : 0.f);
  }
  for (int i = tid; i < kEP * kFF; i += BLOCK) {
    int e = i >> 8, f = i & 255;
    sW2[i] = (_Float16)((e < kE) ? lin2_w[e * kFF + f] : 0.f);
  }
  for (int i = tid; i < kQKVP; i += BLOCK) sBqkv[i] = (i < 84) ? in_proj_b[i] : 0.f;
  for (int i = tid; i < kFF; i += BLOCK) sB1[i] = lin1_b[i];
  if (tid < kEP) {
    const bool v = tid < kE;
    sBout[tid] = v ? out_proj_b[tid] : 0.f;
    sB2[tid]   = v ? lin2_b[tid] : 0.f;
    sLn1g[tid] = v ? ln1_g[tid] : 0.f;  sLn1b[tid] = v ? ln1_b[tid] : 0.f;
    sLn2g[tid] = v ? ln2_g[tid] : 0.f;  sLn2b[tid] = v ? ln2_b[tid] : 0.f;
    sPad[tid]  = v ? pad_token[tid] : 0.f;
  }
  __syncthreads();

  const int w = tid >> 5, lane = tid & 31;
  const int c = lane & 15, hf = lane >> 4;
  _Float16* myT   = sT[w];
  _Float16* myQKV = sQKV[w];

  const int p  = blockIdx.x * NW + w;   // pair index (exact cover, no guard)
  const int b  = p >> 8;                // T/2 == 256 pairs per batch row
  const int t0 = (p & 255) * 2;

  const int len0 = span_lengths[b * kT + t0];
  const int len1 = span_lengths[b * kT + t0 + 1];

  const v8f zero8 = {0.f, 0.f, 0.f, 0.f, 0.f, 0.f, 0.f, 0.f};

  // ---- step 1: load x (16 rows x 28), keep f32 in C layout, stage f16 ----
  const float* px = emb + ((size_t)b * kS + (size_t)t0 * kLMAX) * kE;
  float xr0[8], xr1[8];
#pragma unroll
  for (int r = 0; r < 8; ++r) {
    const int m = r + 8 * hf;
    xr0[r] = px[m * kE + c];
    xr1[r] = (c < 12) ? px[m * kE + 16 + c] : 0.f;
    myT[m * kEP + c]      = (_Float16)xr0[r];
    myT[m * kEP + 16 + c] = (_Float16)xr1[r];
  }
  lds_fence();
  const v16h ax = frag_a(myT, lane);

  // ---- step 2: QKV = x @ Wqkv^T + b, 6 N-tiles of 16 ----
#pragma unroll
  for (int j = 0; j < 6; ++j) {
    const v16h bw  = frag_b(sWqkv + (j * 16 + c) * kEP, hf);
    const v8f acc  = wmma_f16(ax, bw, zero8);
    const float bi = sBqkv[j * 16 + c];
#pragma unroll
    for (int r = 0; r < 8; ++r)
      myQKV[(r + 8 * hf) * kQKVP + j * 16 + c] = (_Float16)(acc[r] + bi);
  }
  lds_fence();

  // ---- step 3: per-head 8x8 attention (VALU), write ao f16 into myT ----
  if (lane < 16) {
#pragma unroll
    for (int jj = 0; jj < 4; ++jj) myT[lane * kEP + kE + jj] = (_Float16)0.f;
  }
#pragma unroll
  for (int rep = 0; rep < 2; ++rep) {
    const int cidx = lane + 32 * rep;      // 64 combos = 16 q-rows x 4 heads
    const int qp = cidx & 15;              // row in the 16-row pair tile
    const int hh = cidx >> 4;              // head 0..3
    const int s  = qp >> 3;                // span within pair
    const int len = s ? len1 : len0;
    const int qoff = hh * kHD;
    float q[7];
#pragma unroll
    for (int d = 0; d < 7; ++d) q[d] = (float)myQKV[qp * kQKVP + qoff + d];
    float sc[8];
    float mx = -1e30f;
#pragma unroll
    for (int k = 0; k < 8; ++k) {
      float dot = 0.f;
#pragma unroll
      for (int d = 0; d < 7; ++d)
        dot += q[d] * (float)myQKV[(s * 8 + k) * kQKVP + kE + qoff + d];
      const float val = (k < len) ? dot * 0.3779644730092272f : -1.0e9f;
      sc[k] = val;
      mx = fmaxf(mx, val);
    }
    float sum = 0.f;
#pragma unroll
    for (int k = 0; k < 8; ++k) { sc[k] = __expf(sc[k] - mx); sum += sc[k]; }
    const float inv = 1.f / sum;
    float ao[7] = {0.f, 0.f, 0.f, 0.f, 0.f, 0.f, 0.f};
#pragma unroll
    for (int k = 0; k < 8; ++k) {
      const float pw = sc[k];
#pragma unroll
      for (int d = 0; d < 7; ++d)
        ao[d] += pw * (float)myQKV[(s * 8 + k) * kQKVP + 2 * kE + qoff + d];
    }
#pragma unroll
    for (int d = 0; d < 7; ++d)
      myT[qp * kEP + qoff + d] = (_Float16)(ao[d] * inv);
  }
  lds_fence();

  // ---- step 4: out-proj (2 N-tiles) + residual + LN1 ----
  const v16h aao = frag_a(myT, lane);
  const v8f o0 = wmma_f16(aao, frag_b(sWout + c * kEP, hf), zero8);
  const v8f o1 = wmma_f16(aao, frag_b(sWout + (16 + c) * kEP, hf), zero8);
  float y0[8], y1[8], z0[8], z1[8];
  {
    const float b0 = sBout[c], b1 = sBout[16 + c];
#pragma unroll
    for (int r = 0; r < 8; ++r) {
      y0[r] = xr0[r] + o0[r] + b0;
      y1[r] = xr1[r] + o1[r] + b1;  // padded cols stay exactly 0
    }
  }
  layer_norm28(y0, y1, sLn1g, sLn1b, c, z0, z1);
#pragma unroll
  for (int r = 0; r < 8; ++r) {
    const int m = r + 8 * hf;
    myT[m * kEP + c]      = (_Float16)z0[r];
    myT[m * kEP + 16 + c] = (_Float16)z1[r];
    xr0[r] = z0[r];
    xr1[r] = z1[r];
  }
  lds_fence();

  // ---- step 5: FF: relu(x@W1^T+b1) @ W2^T + b2, fused over 8 K-chunks ----
  const v16h a2 = frag_a(myT, lane);
  v8f f20 = zero8, f21 = zero8;
#pragma unroll
  for (int i = 0; i < 8; ++i) {
#pragma unroll
    for (int jj = 0; jj < 2; ++jj) {
      const int j = 2 * i + jj;
      const v16h bw1 = frag_b(sW1 + (j * 16 + c) * kEP, hf);
      const v8f hacc = wmma_f16(a2, bw1, zero8);
      const float bb = sB1[j * 16 + c];
#pragma unroll
      for (int r = 0; r < 8; ++r) {
        float hv = hacc[r] + bb;
        hv = hv > 0.f ? hv : 0.f;
        myT[(r + 8 * hf) * kEP + jj * 16 + c] = (_Float16)hv;
      }
    }
    lds_fence();
    const v16h ah = frag_a(myT, lane);
    f20 = wmma_f16(ah, frag_b(sW2 + c * kFF + 32 * i, hf), f20);
    f21 = wmma_f16(ah, frag_b(sW2 + (16 + c) * kFF + 32 * i, hf), f21);
  }
  {
    const float b0 = sB2[c], b1 = sB2[16 + c];
#pragma unroll
    for (int r = 0; r < 8; ++r) {
      y0[r] = xr0[r] + f20[r] + b0;
      y1[r] = xr1[r] + f21[r] + b1;
    }
  }
  layer_norm28(y0, y1, sLn2g, sLn2b, c, z0, z1);

  // ---- step 6: masked mean-pool; lane's half-wave owns span hf ----
  const int t = t0 + hf;
  const int len = hf ? len1 : len0;
  float pooled0 = 0.f, pooled1 = 0.f;
#pragma unroll
  for (int r = 0; r < 8; ++r) {
    if (r < len) { pooled0 += z0[r]; pooled1 += z1[r]; }
  }
  const float invLen = 1.f / (float)len;
  const bool valid = t < num_spans[b];
  float* po = out + ((size_t)b * kT + t) * kE;
  po[c] = valid ? pooled0 * invLen : sPad[c];
  if (c < 12) po[16 + c] = valid ? pooled1 * invLen : sPad[16 + c];
}

extern "C" void kernel_launch(void* const* d_in, const int* in_sizes, int n_in,
                              void* d_out, int out_size, void* d_ws,
                              size_t ws_size, hipStream_t stream) {
  const float* emb          = (const float*)d_in[0];
  const int*   span_lengths = (const int*)d_in[1];
  const int*   num_spans    = (const int*)d_in[2];
  const float* in_proj_w    = (const float*)d_in[3];
  const float* in_proj_b    = (const float*)d_in[4];
  const float* out_proj_w   = (const float*)d_in[5];
  const float* out_proj_b   = (const float*)d_in[6];
  const float* ln1_g        = (const float*)d_in[7];
  const float* ln1_b        = (const float*)d_in[8];
  const float* lin1_w       = (const float*)d_in[9];
  const float* lin1_b       = (const float*)d_in[10];
  const float* lin2_w       = (const float*)d_in[11];
  const float* lin2_b       = (const float*)d_in[12];
  const float* ln2_g        = (const float*)d_in[13];
  const float* ln2_b        = (const float*)d_in[14];
  const float* pad_token    = (const float*)d_in[15];
  float* out = (float*)d_out;

  hipLaunchKernelGGL(span_encoder_kernel, dim3(GRID), dim3(BLOCK), 0, stream,
                     emb, span_lengths, num_spans, in_proj_w, in_proj_b,
                     out_proj_w, out_proj_b, ln1_g, ln1_b, lin1_w, lin1_b,
                     lin2_w, lin2_b, ln2_g, ln2_b, pad_token, out);
}
